// SSD300_16664473108733
// MI455X (gfx1250) — compile-verified, matrix operand order checked by reading, not run on hardware
//
#include <hip/hip_runtime.h>
#include <hip/hip_bf16.h>
#include <stdint.h>

// ---------------- problem constants (from reference) ----------------
constexpr int kB   = 32;
constexpr int kN   = 8732;
constexpr int kCLS = 81;
constexpr int kC   = 80;          // CLS - 1
constexpr int kK   = 200;
constexpr float kDetThr = 0.05f;
constexpr float kIouThr = 0.5f;

constexpr int kTile  = 128;       // n-tile for softmax pass (TDM tile_dim0)
constexpr int kMaxC  = 2048;      // candidate cap per (b,c) row
constexpr int kKPad  = 208;       // K padded to 13*16 for WMMA tiles
constexpr int kTiles = 13;        // 208/16

constexpr size_t kBKC  = (size_t)kB * kK * kC;        // 512000
constexpr size_t kOff1 = kBKC * 4;                    // probs
constexpr size_t kOff2 = kOff1 + kBKC;                // cls
constexpr size_t kOff3 = kOff2 + kBKC;                // img
constexpr size_t kOff4 = kOff3 + kBKC;                // keep

typedef __attribute__((ext_vector_type(4)))  unsigned int v4u;
typedef __attribute__((ext_vector_type(8)))  int          v8i;
typedef __attribute__((ext_vector_type(4)))  int          v4i;
typedef __attribute__((ext_vector_type(16))) _Float16     v16h;
typedef __attribute__((ext_vector_type(8)))  float        v8f;

__device__ __forceinline__ unsigned ballot32(int pred) {
#if __has_builtin(__builtin_amdgcn_ballot_w32)
  return __builtin_amdgcn_ballot_w32(pred != 0);
#else
  return (unsigned)__ballot(pred);   // wave32: low 32 bits
#endif
}

// =====================================================================
// Kernel 1: per-(b,n) softmax stats (max, 1/sum) + box decode.
// Scores tile (81 classes x 128 n) is DMA'd into LDS by the Tensor Data
// Mover (async tensor path, TENSORcnt), then each thread reduces its
// n-column.  Grid: (ceil(N/128), B), block: 128.
// =====================================================================
__global__ __launch_bounds__(kTile)
void ssd_softmax_decode(const float* __restrict__ locs,
                        const float* __restrict__ scores,
                        const float* __restrict__ dboxes,
                        float* __restrict__ mx,
                        float* __restrict__ ism,
                        float* __restrict__ boxes) {
  __shared__ float sm[kCLS * kTile];           // 41,472 B
  const int b   = blockIdx.y;
  const int n0  = blockIdx.x * kTile;
  const int tid = threadIdx.x;

#if defined(__AMDGCN__) && __has_builtin(__builtin_amdgcn_tensor_load_to_lds)
  if (tid < 32) {  // wave 0 issues the TDM descriptor (EXEC ignored by TDM)
    unsigned long long ga =
        (unsigned long long)(uintptr_t)(scores + ((size_t)b * kCLS) * kN + n0);
    unsigned ldsb = (unsigned)(uintptr_t)&sm[0];   // LDS byte offset (addr[31:0])
    v4u g0;
    g0[0] = 1u;                                    // count=1, user mode
    g0[1] = ldsb;                                  // lds_addr
    g0[2] = (unsigned)(ga & 0xFFFFFFFFu);          // global_addr[31:0]
    g0[3] = (unsigned)((ga >> 32) & 0x01FFFFFFu)   // global_addr[56:32]
          | 0x80000000u;                           // type=2 ("image")
    v8i g1;
    g1[0] = (int)(2u << 16);                               // data_size=4B
    g1[1] = (int)(((unsigned)kN & 0xFFFFu) << 16);         // tensor_dim0 lo16
    g1[2] = (int)((((unsigned)kN >> 16) & 0xFFFFu)         // tensor_dim0 hi16
          | (((unsigned)kCLS & 0xFFFFu) << 16));           // tensor_dim1 lo16
    g1[3] = (int)(((unsigned)kTile & 0xFFFFu) << 16);      // tile_dim0=128
    g1[4] = (int)((unsigned)kCLS & 0xFFFFu);               // tile_dim1=81
    g1[5] = (int)(unsigned)kN;                             // dim0_stride lo32
    g1[6] = 0;
    g1[7] = 0;
    v4i g2 = {}; v4i g3 = {};
    v8i g4 = {};                                   // unused trailing group
    __builtin_amdgcn_tensor_load_to_lds(g0, g1, g2, g3, g4, 0);
    __builtin_amdgcn_s_wait_tensorcnt(0);
  }
#else
  for (int idx = tid; idx < kCLS * kTile; idx += kTile) {
    const int c  = idx / kTile;
    const int nn = n0 + (idx % kTile);
    sm[idx] = (nn < kN) ? scores[((size_t)b * kCLS + c) * kN + nn] : 0.0f;
  }
#endif
  __syncthreads();

  const int n = n0 + tid;
  float m = -1e30f;
  for (int c = 0; c < kCLS; ++c) m = fmaxf(m, sm[c * kTile + tid]);
  float s = 0.0f;
  for (int c = 0; c < kCLS; ++c) s += __expf(sm[c * kTile + tid] - m);

  if (n < kN) {
    mx [(size_t)b * kN + n] = m;
    ism[(size_t)b * kN + n] = 1.0f / s;

    // box decode (normalized xyxy)
    const float lx = locs[((size_t)b * 4 + 0) * kN + n] * 0.1f;
    const float ly = locs[((size_t)b * 4 + 1) * kN + n] * 0.1f;
    const float lw = locs[((size_t)b * 4 + 2) * kN + n] * 0.2f;
    const float lh = locs[((size_t)b * 4 + 3) * kN + n] * 0.2f;
    const float4 d = ((const float4*)dboxes)[n];   // (cx, cy, w, h)
    const float cx = lx * d.z + d.x;
    const float cy = ly * d.w + d.y;
    const float w  = __expf(lw) * d.z;
    const float h  = __expf(lh) * d.w;
    float4 bx;
    bx.x = cx - 0.5f * w; bx.y = cy - 0.5f * h;
    bx.z = cx + 0.5f * w; bx.w = cy + 0.5f * h;
    ((float4*)boxes)[(size_t)b * kN + n] = bx;
  }
}

// =====================================================================
// Kernel 2: per-(b,class) top-K.  Only probs > DET_THR can survive the
// reference's keep-gating, so filter candidates first (scores re-read is
// L2-resident: 90MB < 192MB L2), then K arg-max selections over the
// small candidate set.  Intra-wave reduction via wave32 shuffles
// (ds_permute, no barriers); only the 8 wave partials cross LDS.
// Ties broken by smaller index (top_k semantics).  Grid: B*C x 256.
// =====================================================================
__global__ __launch_bounds__(256)
void ssd_topk(const float* __restrict__ scores,
              const float* __restrict__ mx,
              const float* __restrict__ ism,
              float* __restrict__ topv,
              int*   __restrict__ topi) {
  __shared__ float candv[kMaxC];
  __shared__ int   candi[kMaxC];
  __shared__ int   cnt;
  __shared__ float rv[8];
  __shared__ int   ri[8];
  __shared__ int   rp[8];
  __shared__ int   stopk;

  const int row  = blockIdx.x;           // b*C + cc
  const int b    = row / kC;
  const int cc   = row % kC;
  const int cls  = cc + 1;               // softmax class (class 0 dropped)
  const int tid  = threadIdx.x;
  const int wave = tid >> 5, lane = tid & 31;

  if (tid == 0) { cnt = 0; stopk = kK; }
  __syncthreads();

  const float* srow = scores + ((size_t)b * kCLS + cls) * kN;
  const float* mrow = mx  + (size_t)b * kN;
  const float* irow = ism + (size_t)b * kN;
  for (int n = tid; n < kN; n += 256) {
    const float p = __expf(srow[n] - mrow[n]) * irow[n];
    if (p > kDetThr) {
      const int pos = atomicAdd(&cnt, 1);
      if (pos < kMaxC) { candv[pos] = p; candi[pos] = n; }
    }
  }
  __syncthreads();
  const int m = (cnt < kMaxC) ? cnt : kMaxC;

  for (int k = 0; k < kK; ++k) {
    float bv = -1.0f; int bi = 0x7FFFFFFF; int bp = -1;
    for (int p = tid; p < m; p += 256) {
      const float v = candv[p]; const int i = candi[p];
      if (v > bv || (v == bv && i < bi)) { bv = v; bi = i; bp = p; }
    }
    // wave32 butterfly reduction (lane permutes, no LDS/barriers)
    for (int off = 16; off > 0; off >>= 1) {
      const float ov = __shfl_xor(bv, off, 32);
      const int   oi = __shfl_xor(bi, off, 32);
      const int   op = __shfl_xor(bp, off, 32);
      if (ov > bv || (ov == bv && oi < bi)) { bv = ov; bi = oi; bp = op; }
    }
    if (lane == 0) { rv[wave] = bv; ri[wave] = bi; rp[wave] = bp; }
    __syncthreads();
    if (tid == 0) {
      float fv = rv[0]; int fi = ri[0]; int fp = rp[0];
      for (int w = 1; w < 8; ++w) {
        if (rv[w] > fv || (rv[w] == fv && ri[w] < fi)) {
          fv = rv[w]; fi = ri[w]; fp = rp[w];
        }
      }
      if (fv > 0.0f) {
        topv[(size_t)row * kK + k] = fv;
        topi[(size_t)row * kK + k] = fi;
        candv[fp] = -1.0f;                  // remove winner
      } else {
        stopk = k;                          // candidates exhausted
      }
    }
    __syncthreads();
    if (k >= stopk) break;
  }
  for (int k = stopk + tid; k < kK; k += 256) {   // sub-threshold slots -> 0
    topv[(size_t)row * kK + k] = 0.0f;
    topi[(size_t)row * kK + k] = 0;
  }
}

// =====================================================================
// Kernel 3: bitmask NMS + output assembly.  The pairwise IOU denominator
// (area_i + area_j) is produced 256-pairs-at-a-time with
// v_wmma_f32_16x16x32_f16: A row i = [a_i, 1, 0...], B col j = [1, a_j,
// 0...] (only K=0,1 populated -> fragment halves in lanes 0-15 per the
// §7.12.2 layouts).  IOU test is division-free:
//   inter/(union+eps) > thr  <=>  inter > thr*(union+eps)   (union+eps>0)
// IOU on normalized coords (scale-invariant).  Suppression bits packed
// via wave32 ballot; greedy scan by one lane.  Grid: B*C blocks of 256.
// =====================================================================
__global__ __launch_bounds__(256)
void ssd_nms_out(const float* __restrict__ topv,
                 const int*   __restrict__ topi,
                 const float* __restrict__ boxes,
                 const float* __restrict__ heights,
                 const float* __restrict__ widths,
                 float* __restrict__ out) {
  __shared__ float    x1L[kKPad], y1L[kKPad], x2L[kKPad], y2L[kKPad], areaL[kKPad];
  __shared__ unsigned supL[kKPad * 7];
  __shared__ unsigned vwL[7];

  const int row = blockIdx.x;
  const int b   = row / kC;
  const int cc  = row % kC;
  const int tid = threadIdx.x;
  const float* tv = topv + (size_t)row * kK;
  const int*   ti = topi + (size_t)row * kK;
  const float4* bptr = (const float4*)boxes;

  for (int w = tid; w < kKPad * 7; w += 256) supL[w] = 0u;
  if (tid < 7) vwL[tid] = 0u;
  __syncthreads();

  if (tid < kKPad) {
    float a = 0.f, bb = 0.f, c2 = 0.f, d2 = 0.f;
    if (tid < kK) {
      const float v  = tv[tid];
      const int  idx = ti[tid];
      const float4 bx = bptr[(size_t)b * kN + idx];
      a = bx.x; bb = bx.y; c2 = bx.z; d2 = bx.w;
      if (v > kDetThr) atomicOr(&vwL[tid >> 5], 1u << (tid & 31));
    }
    x1L[tid] = a;  y1L[tid] = bb; x2L[tid] = c2; y2L[tid] = d2;
    areaL[tid] = (c2 - a) * (d2 - bb);
  }
  __syncthreads();

  const int wave = tid >> 5, lane = tid & 31;
  const int half = lane >> 4, l15 = lane & 15;
  for (int t = wave; t < kTiles * kTiles; t += 8) {
    const int i0 = (t / kTiles) * 16;
    const int j0 = (t % kTiles) * 16;

    const float ai = half ? 0.0f : areaL[i0 + l15];
    const float aj = half ? 0.0f : areaL[j0 + l15];
    v16h Af = {}; v16h Bf = {};
    Af[0] = (_Float16)ai;
    Af[1] = half ? (_Float16)0.0f : (_Float16)1.0f;
    Bf[0] = half ? (_Float16)0.0f : (_Float16)1.0f;
    Bf[1] = (_Float16)aj;
    v8f Cz = {};
    v8f S;
#if defined(__AMDGCN__) && __has_builtin(__builtin_amdgcn_wmma_f32_16x16x32_f16)
    // D[M][N] = a_{i0+M} + a_{j0+N}
    S = __builtin_amdgcn_wmma_f32_16x16x32_f16(false, Af, false, Bf,
                                               (short)0, Cz, false, false);
#else
    for (int r = 0; r < 8; ++r) S[r] = areaL[i0 + r + half * 8] + areaL[j0 + l15];
    (void)Cz;
#endif

    const int j = j0 + l15;
    const float jx1 = x1L[j], jy1 = y1L[j], jx2 = x2L[j], jy2 = y2L[j];
    for (int r = 0; r < 8; ++r) {
      const int i = i0 + r + (half << 3);
      const float iwd = fminf(x2L[i], jx2) - fmaxf(x1L[i], jx1);
      const float ihd = fminf(y2L[i], jy2) - fmaxf(y1L[i], jy1);
      const float inter = fmaxf(iwd, 0.f) * fmaxf(ihd, 0.f);
      const float uni = S[r] - inter + 1e-9f;     // union + eps, always > 0
      const int pred = (inter > kIouThr * uni) && (j > i) && (j < kK) && (i < kK);
      const unsigned msk = ballot32(pred);
      if (lane == 0) {
        const unsigned lo = msk & 0xFFFFu, hi = msk >> 16;
        const int word = j0 >> 5, sh = j0 & 31;
        if (lo) atomicOr(&supL[(i0 + r)     * 7 + word], lo << sh);
        if (hi) atomicOr(&supL[(i0 + 8 + r) * 7 + word], hi << sh);
      }
    }
  }
  __syncthreads();

  if (tid == 0) {   // greedy bitmask scan (score order == k order)
    for (int i = 0; i < kK; ++i) {
      if ((vwL[i >> 5] >> (i & 31)) & 1u) {
        for (int w = 0; w < 7; ++w) vwL[w] &= ~supL[i * 7 + w];
      }
    }
  }
  __syncthreads();

  const float Wimg = widths[b], Himg = heights[b];
  for (int k = tid; k < kK; k += 256) {
    const int keep = (vwL[k >> 5] >> (k & 31)) & 1u;
    const float v  = tv[k];
    const int  idx = ti[k];
    const float4 bx = bptr[(size_t)b * kN + idx];
    const float x1p = bx.x * Wimg, y1p = bx.y * Himg;
    const float x2p = bx.z * Wimg, y2p = bx.w * Himg;
    const size_t orow = ((size_t)b * kK + k) * kC + cc;
    if (keep) {
      out[orow * 4 + 0] = x1p;
      out[orow * 4 + 1] = y1p;
      out[orow * 4 + 2] = x2p - x1p;
      out[orow * 4 + 3] = y2p - y1p;
      out[kOff1 + orow] = v;
    } else {
      out[orow * 4 + 0] = 0.f; out[orow * 4 + 1] = 0.f;
      out[orow * 4 + 2] = 0.f; out[orow * 4 + 3] = 0.f;
      out[kOff1 + orow] = 0.f;
    }
    out[kOff2 + orow] = (float)(cc + 1);
    out[kOff3 + orow] = (float)b;
    out[kOff4 + orow] = (float)keep;
  }
}

// =====================================================================
extern "C" void kernel_launch(void* const* d_in, const int* in_sizes, int n_in,
                              void* d_out, int out_size, void* d_ws, size_t ws_size,
                              hipStream_t stream) {
  const float* locs    = (const float*)d_in[0];   // (B,4,N)
  const float* scores  = (const float*)d_in[1];   // (B,CLS,N)
  const float* heights = (const float*)d_in[2];   // (B,)
  const float* widths  = (const float*)d_in[3];   // (B,)
  const float* dboxes  = (const float*)d_in[4];   // (N,4)

  char* ws = (char*)d_ws;
  size_t off = 0;
  float* boxes = (float*)(ws + off); off += (size_t)kB * kN * 4 * sizeof(float);
  float* mx    = (float*)(ws + off); off += (size_t)kB * kN * sizeof(float);
  float* ism   = (float*)(ws + off); off += (size_t)kB * kN * sizeof(float);
  float* topv  = (float*)(ws + off); off += (size_t)kB * kC * kK * sizeof(float);
  int*   topi  = (int*)  (ws + off); off += (size_t)kB * kC * kK * sizeof(int);
  (void)ws_size; (void)in_sizes; (void)n_in; (void)out_size;

  dim3 g1((kN + kTile - 1) / kTile, kB);
  ssd_softmax_decode<<<g1, kTile, 0, stream>>>(locs, scores, dboxes, mx, ism, boxes);
  ssd_topk<<<kB * kC, 256, 0, stream>>>(scores, mx, ism, topv, topi);
  ssd_nms_out<<<kB * kC, 256, 0, stream>>>(topv, topi, boxes, heights, widths,
                                           (float*)d_out);
}
